// DGLJTNNDecoder_39960375722853
// MI455X (gfx1250) — compile-verified, matrix operand order checked by reading, not run on hardware
//
#include <hip/hip_runtime.h>
#include <hip/hip_bf16.h>
#include <math.h>

// ---------------- problem constants (match reference) ----------------
#define BB 512          // trees
#define NN 48           // nodes per tree
#define HH 256          // hidden
#define LL 64           // latent
#define VV 800          // vocab
#define TT (2*(NN-1))   // 94 sequential steps
#define ETOT (BB*TT)    // 48128 edges total
#define NFWD (NN-1)     // 47
#define QROWS ((NFWD+1)*BB)   // 24576
#define PROWS ((TT+1)*BB)     // 48640
#define PTROWS (TT*BB)        // 48128

typedef __attribute__((ext_vector_type(16))) _Float16 v16h;
typedef __attribute__((ext_vector_type(8)))  float    v8f;

// ---------------- WMMA fragment helpers (gfx1250, wave32) ----------------
// 16-bit A-matrix 16x32 layout: lane l holds row M = l%16; element e of v16h
// maps to K = ((e>>3)<<4) + (((e>>1)&3)<<1) + (e&1) + 8*(l/16).
__device__ __forceinline__ int ke_off(int e) {
  return ((e >> 3) << 4) + (((e >> 1) & 3) << 1) + (e & 1);
}

__device__ __forceinline__ v16h zero_v16h() {
  v16h a;
#pragma unroll
  for (int e = 0; e < 16; ++e) a[e] = (_Float16)0.0f;
  return a;
}

// A tile: 16x32 block of row-major f16 matrix at (row, k0); per-lane K offsets
// are contiguous bytes -> 2x global_load_b128.
__device__ __forceinline__ v16h load_a_tile(const _Float16* __restrict__ A, long lda,
                                            int row, int k0, int lane) {
  v16h a;
  const _Float16* p = A + (long)row * lda + k0 + ((lane >> 4) << 3);
#pragma unroll
  for (int e = 0; e < 16; ++e) a[e] = p[ke_off(e)];
  return a;
}

// Packed B: fragment-native layout [ktile][ntile][lane][16]; one aligned
// 32-byte v16h load per lane.
__device__ __forceinline__ v16h load_b_packed(const _Float16* __restrict__ Bp,
                                              int ntile, int kt, int nt, int lane) {
  return ((const v16h*)Bp)[((kt * ntile + nt) << 5) + lane];
}

// Pack row-major f32 W[K][N] -> WMMA-native f16 fragments.
__global__ void k_pack_b(_Float16* __restrict__ dst, const float* __restrict__ src,
                         int K, int N) {
  int idx = blockIdx.x * blockDim.x + threadIdx.x;  // one thread per (tile, lane)
  int total = (K >> 5) * (N >> 4) * 32;
  if (idx >= total) return;
  int lane = idx & 31;
  int tile = idx >> 5;
  int ntile = N >> 4;
  int kt = tile / ntile, nt = tile % ntile;
  int n = (nt << 4) + (lane & 15);
  int kbase = (kt << 5) + ((lane >> 4) << 3);
  _Float16* d = dst + ((size_t)idx << 4);
#pragma unroll
  for (int e = 0; e < 16; ++e)
    d[e] = (_Float16)src[(size_t)(kbase + ke_off(e)) * N + n];
}

// ---------------- WMMA GEMM, 32x32 macro-tile per wave ----------------
// 4 independent accumulator chains per K-step hide WMMA hazards; A/B fragments
// reused 2x each. C = act(A@Bpacked + bias).
__global__ void k_gemm(const _Float16* __restrict__ A, int lda,
                       const _Float16* __restrict__ Bp,
                       const float* __restrict__ bias,
                       float* __restrict__ Cf32, _Float16* __restrict__ Cf16, int ldc,
                       int M, int N, int K, int act) {
  int lane = threadIdx.x & 31;
  int w = (int)((blockIdx.x * blockDim.x + threadIdx.x) >> 5);
  int ntile2 = N >> 5;                       // 32-wide column tiles
  if (w >= (M >> 5) * ntile2) return;
  int tm = w / ntile2, tn = w % ntile2;
  int m0 = tm << 5, n0 = tn << 5;
  int ntile = N >> 4;                        // 16-wide tiles in packed B
  int nt0 = tn << 1, nt1 = nt0 + 1;
  v8f c00 = {0.f,0.f,0.f,0.f,0.f,0.f,0.f,0.f};
  v8f c01 = c00, c10 = c00, c11 = c00;
  int ar0 = m0 + (lane & 15);
  int ar1 = ar0 + 16;
  int kt = 0;
  for (int k0 = 0; k0 < K; k0 += 32, ++kt) {
    v16h a0 = load_a_tile(A, lda, ar0, k0, lane);
    v16h a1 = load_a_tile(A, lda, ar1, k0, lane);
    v16h b0 = load_b_packed(Bp, ntile, kt, nt0, lane);
    v16h b1 = load_b_packed(Bp, ntile, kt, nt1, lane);
    c00 = __builtin_amdgcn_wmma_f32_16x16x32_f16(false, a0, false, b0, (short)0, c00, false, false);
    c01 = __builtin_amdgcn_wmma_f32_16x16x32_f16(false, a0, false, b1, (short)0, c01, false, false);
    c10 = __builtin_amdgcn_wmma_f32_16x16x32_f16(false, a1, false, b0, (short)0, c10, false, false);
    c11 = __builtin_amdgcn_wmma_f32_16x16x32_f16(false, a1, false, b1, (short)0, c11, false, false);
  }
  int nA = n0 + (lane & 15), nB = nA + 16;
  float bvA = bias ? bias[nA] : 0.0f;
  float bvB = bias ? bias[nB] : 0.0f;
  int mbase = m0 + ((lane >> 4) << 3);
#pragma unroll
  for (int j = 0; j < 8; ++j) {
    int mA = mbase + j, mB = mA + 16;
    float v00 = c00[j] + bvA, v01 = c01[j] + bvB;
    float v10 = c10[j] + bvA, v11 = c11[j] + bvB;
    if (act) {
      v00 = fmaxf(v00, 0.f); v01 = fmaxf(v01, 0.f);
      v10 = fmaxf(v10, 0.f); v11 = fmaxf(v11, 0.f);
    }
    if (Cf32) {
      Cf32[(long)mA * ldc + nA] = v00; Cf32[(long)mA * ldc + nB] = v01;
      Cf32[(long)mB * ldc + nA] = v10; Cf32[(long)mB * ldc + nB] = v11;
    }
    if (Cf16) {
      Cf16[(long)mA * ldc + nA] = (_Float16)v00; Cf16[(long)mA * ldc + nB] = (_Float16)v01;
      Cf16[(long)mB * ldc + nA] = (_Float16)v10; Cf16[(long)mB * ldc + nB] = (_Float16)v11;
    }
  }
}

// ---------------- concat-A WMMA GEMM (32x32 tile) with row indirection ----------------
// Virtual A = [seg0 | seg1 | seg2]; Ii[r] < 0 => zero row. Epilogue relu+bias -> f16.
__global__ void k_gemm_concat(const _Float16* __restrict__ A0, const int* __restrict__ I0, int ld0, int K0,
                              const _Float16* __restrict__ A1, const int* __restrict__ I1, int ld1, int K1,
                              const _Float16* __restrict__ A2, const int* __restrict__ I2, int ld2, int K2,
                              const _Float16* __restrict__ Bp,
                              const float* __restrict__ bias,
                              _Float16* __restrict__ Cf16, int ldc,
                              int M, int N, int act) {
  int lane = threadIdx.x & 31;
  int w = (int)((blockIdx.x * blockDim.x + threadIdx.x) >> 5);
  int ntile2 = N >> 5;
  if (w >= (M >> 5) * ntile2) return;
  int tm = w / ntile2, tn = w % ntile2;
  int m0 = tm << 5, n0 = tn << 5;
  int ntile = N >> 4;
  int nt0 = tn << 1, nt1 = nt0 + 1;
  int K = K0 + K1 + K2;
  v8f c00 = {0.f,0.f,0.f,0.f,0.f,0.f,0.f,0.f};
  v8f c01 = c00, c10 = c00, c11 = c00;
  int r0 = m0 + (lane & 15);
  int r1 = r0 + 16;
  int kt = 0;
  for (int k0 = 0; k0 < K; k0 += 32, ++kt) {
    const _Float16* Ap; const int* Ip; int ld; int kl;
    if (k0 < K0)            { Ap = A0; Ip = I0; ld = ld0; kl = k0; }
    else if (k0 < K0 + K1)  { Ap = A1; Ip = I1; ld = ld1; kl = k0 - K0; }
    else                    { Ap = A2; Ip = I2; ld = ld2; kl = k0 - K0 - K1; }
    int ar0 = Ip ? Ip[r0] : r0;
    int ar1 = Ip ? Ip[r1] : r1;
    v16h a0 = (ar0 < 0) ? zero_v16h() : load_a_tile(Ap, ld, ar0, kl, lane);
    v16h a1 = (ar1 < 0) ? zero_v16h() : load_a_tile(Ap, ld, ar1, kl, lane);
    v16h b0 = load_b_packed(Bp, ntile, kt, nt0, lane);
    v16h b1 = load_b_packed(Bp, ntile, kt, nt1, lane);
    c00 = __builtin_amdgcn_wmma_f32_16x16x32_f16(false, a0, false, b0, (short)0, c00, false, false);
    c01 = __builtin_amdgcn_wmma_f32_16x16x32_f16(false, a0, false, b1, (short)0, c01, false, false);
    c10 = __builtin_amdgcn_wmma_f32_16x16x32_f16(false, a1, false, b0, (short)0, c10, false, false);
    c11 = __builtin_amdgcn_wmma_f32_16x16x32_f16(false, a1, false, b1, (short)0, c11, false, false);
  }
  int nA = n0 + (lane & 15), nB = nA + 16;
  float bvA = bias ? bias[nA] : 0.0f;
  float bvB = bias ? bias[nB] : 0.0f;
  int mbase = m0 + ((lane >> 4) << 3);
#pragma unroll
  for (int j = 0; j < 8; ++j) {
    int mA = mbase + j, mB = mA + 16;
    float v00 = c00[j] + bvA, v01 = c01[j] + bvB;
    float v10 = c10[j] + bvA, v11 = c11[j] + bvB;
    if (act) {
      v00 = fmaxf(v00, 0.f); v01 = fmaxf(v01, 0.f);
      v10 = fmaxf(v10, 0.f); v11 = fmaxf(v11, 0.f);
    }
    Cf16[(long)mA * ldc + nA] = (_Float16)v00; Cf16[(long)mA * ldc + nB] = (_Float16)v01;
    Cf16[(long)mB * ldc + nA] = (_Float16)v10; Cf16[(long)mB * ldc + nB] = (_Float16)v11;
  }
}

// ---------------- small utility kernels ----------------
__global__ void k_zero_f32(float* p, int n) {
  int i = blockIdx.x * blockDim.x + threadIdx.x;
  if (i < n) p[i] = 0.0f;
}
__global__ void k_zero_h(_Float16* p, int n) {
  int i = blockIdx.x * blockDim.x + threadIdx.x;
  if (i < n) p[i] = (_Float16)0.0f;
}
__global__ void k_cvt(_Float16* dst, const float* src, int n) {
  int i = blockIdx.x * blockDim.x + threadIdx.x;
  if (i < n) dst[i] = (_Float16)src[i];
}
__global__ void k_embed(const float* __restrict__ emb, const int* __restrict__ wid,
                        _Float16* __restrict__ x, int n) {
  int i = blockIdx.x * blockDim.x + threadIdx.x;
  if (i >= n) return;
  int node = i >> 8, h = i & 255;
  x[i] = (_Float16)emb[(long)wid[node] * HH + h];
}
__global__ void k_build_idx(const int* __restrict__ root_ids, const int* __restrict__ step_v,
                            int* pnode, int* ph, int* pb, int* qh) {
  int r = blockIdx.x * blockDim.x + threadIdx.x;
  if (r >= PROWS) return;
  int b = r & (BB - 1);
  pb[r] = b;
  pnode[r] = (r < BB) ? root_ids[b] : step_v[r - BB];
  ph[r] = (r < BB) ? -1 : r - BB;
  if (r < QROWS) qh[r] = (r < BB) ? -1 : r - BB;
}

// ---------------- per-step kernels ----------------
__global__ void k_step_prepare(const int* __restrict__ eid, const int* __restrict__ pred,
                               const unsigned char* __restrict__ pmask,
                               const int* __restrict__ edge_src,
                               const _Float16* __restrict__ x,
                               const _Float16* __restrict__ m16, const _Float16* __restrict__ rm16,
                               _Float16* __restrict__ A1, _Float16* __restrict__ A2) {
  int i = blockIdx.x * blockDim.x + threadIdx.x;
  if (i >= BB * HH) return;
  int b = i >> 8, h = i & 255;
  int e = eid[b];
  _Float16 sx = x[(long)edge_src[e] * HH + h];
  _Float16 s = (_Float16)0.0f, arm = (_Float16)0.0f;
  if (pmask[b]) {
    int p = pred[b];
    s   = m16[(long)p * HH + h];
    arm = rm16[(long)p * HH + h];
  }
  A1[(long)b * 512 + h] = sx; A1[(long)b * 512 + HH + h] = s;
  A2[(long)b * 512 + h] = sx; A2[(long)b * 512 + HH + h] = arm;
}

__global__ void k_step_elem(const float* __restrict__ zpre, const float* __restrict__ mtpre,
                            const float* __restrict__ bz, const float* __restrict__ bh,
                            const _Float16* __restrict__ A1,
                            const int* __restrict__ eid, const int* __restrict__ edge_dst,
                            const _Float16* __restrict__ x,
                            _Float16* __restrict__ m16, _Float16* __restrict__ A3) {
  int i = blockIdx.x * blockDim.x + threadIdx.x;
  if (i >= BB * HH) return;
  int b = i >> 8, h = i & 255;
  float z  = 1.0f / (1.0f + expf(-(zpre[i] + bz[h])));
  float mt = tanhf(mtpre[i] + bh[h]);
  float s  = (float)A1[(long)b * 512 + HH + h];
  float me = (1.0f - z) * s + z * mt;
  int e = eid[b];
  m16[(long)e * HH + h] = (_Float16)me;
  A3[(long)b * 512 + h]      = x[(long)edge_dst[e] * HH + h];
  A3[(long)b * 512 + HH + h] = (_Float16)me;
}

__global__ void k_step_finish(const float* __restrict__ rpre, const float* __restrict__ br,
                              const int* __restrict__ eid,
                              const int* __restrict__ in0, const unsigned char* __restrict__ m0m,
                              const int* __restrict__ in1, const unsigned char* __restrict__ m1m,
                              _Float16* __restrict__ m16, _Float16* __restrict__ rm16,
                              _Float16* __restrict__ hsOut) {
  int i = blockIdx.x * blockDim.x + threadIdx.x;
  if (i >= BB * HH) return;
  int b = i >> 8, h = i & 255;
  float r = 1.0f / (1.0f + expf(-(rpre[i] + br[h])));
  int e = eid[b];
  float me = (float)m16[(long)e * HH + h];
  rm16[(long)e * HH + h] = (_Float16)(r * me);
  float hv = 0.0f;
  if (m0m[b]) hv += (float)m16[(long)in0[b] * HH + h];
  if (m1m[b]) hv += (float)m16[(long)in1[b] * HH + h];
  hsOut[i] = (_Float16)hv;
}

// ---------------- q head: WMMA logits (16x32/wave) + fused log-softmax ----------------
__global__ void k_qsoftmax(const _Float16* __restrict__ relu1q, const _Float16* __restrict__ WoP,
                           const float* __restrict__ Wo_b,
                           const int* __restrict__ wid, const int* __restrict__ root_ids,
                           const int* __restrict__ step_v, float* __restrict__ accum) {
  __shared__ float logits[16][VV];
  int lane = threadIdx.x & 31;
  int wv = threadIdx.x >> 5;        // 0..7
  int m0 = blockIdx.x << 4;
  const int ntile = VV / 16;        // 50 (16-wide), 25 macro tiles of 32
  int arow = m0 + (lane & 15);
  for (int ct2 = wv; ct2 < 25; ct2 += 8) {
    int nt0 = ct2 << 1, nt1 = nt0 + 1;
    v8f c0 = {0.f,0.f,0.f,0.f,0.f,0.f,0.f,0.f};
    v8f c1 = c0;
    int kt = 0;
    for (int k0 = 0; k0 < HH; k0 += 32, ++kt) {
      v16h a  = load_a_tile(relu1q, HH, arow, k0, lane);
      v16h b0 = load_b_packed(WoP, ntile, kt, nt0, lane);
      v16h b1 = load_b_packed(WoP, ntile, kt, nt1, lane);
      c0 = __builtin_amdgcn_wmma_f32_16x16x32_f16(false, a, false, b0, (short)0, c0, false, false);
      c1 = __builtin_amdgcn_wmma_f32_16x16x32_f16(false, a, false, b1, (short)0, c1, false, false);
    }
    int nA = (nt0 << 4) + (lane & 15), nB = nA + 16;
    float bvA = Wo_b[nA], bvB = Wo_b[nB];
    int jm = (lane >> 4) << 3;
#pragma unroll
    for (int j = 0; j < 8; ++j) {
      logits[j + jm][nA] = c0[j] + bvA;
      logits[j + jm][nB] = c1[j] + bvB;
    }
  }
  __syncthreads();
  if (threadIdx.x < 16) {
    int r = m0 + threadIdx.x;
    int ti = r >> 9, b = r & (BB - 1);
    int node = (ti == 0) ? root_ids[b] : step_v[((ti - 1) << 9) + b];
    int tgt = wid[node];
    float mx = -1e30f; int am = 0;
    for (int v = 0; v < VV; ++v) {
      float lv = logits[threadIdx.x][v];
      if (lv > mx) { mx = lv; am = v; }
    }
    float se = 0.0f;
    for (int v = 0; v < VV; ++v) se += expf(logits[threadIdx.x][v] - mx);
    float loss = logf(se) + mx - logits[threadIdx.x][tgt];
    atomicAdd(&accum[0], loss);
    if (am == tgt) atomicAdd(&accum[2], 1.0f);
  }
}

// ---------------- p head: per-row dot + BCE-with-logits ----------------
__global__ void k_ploss(const _Float16* __restrict__ relu1p, const float* __restrict__ Us_w,
                        const float* __restrict__ Us_b, const int* __restrict__ step_label,
                        float* __restrict__ accum) {
  int lane = threadIdx.x & 31;
  int row = (blockIdx.x << 3) + (threadIdx.x >> 5);
  if (row >= PROWS) return;
  float s = 0.0f;
  for (int k = lane; k < HH; k += 32) s += (float)relu1p[(long)row * HH + k] * Us_w[k];
  for (int off = 16; off > 0; off >>= 1) s += __shfl_down(s, off, 32);
  if (lane == 0) {
    float p = s + Us_b[0];
    float tgt = (row < PTROWS) ? (1.0f - (float)step_label[row >> 9]) : 0.0f;
    float loss = fmaxf(p, 0.0f) - p * tgt + log1pf(expf(-fabsf(p)));
    atomicAdd(&accum[1], loss);
    int pred = (p > 0.0f) ? 1 : 0;
    if ((float)pred == tgt) atomicAdd(&accum[3], 1.0f);
  }
}

__global__ void k_final(const float* __restrict__ accum, float* __restrict__ out) {
  if (threadIdx.x == 0 && blockIdx.x == 0) {
    out[0] = accum[0] / (float)BB;      // q_loss
    out[1] = accum[1] / (float)BB;      // p_loss
    out[2] = accum[2] / (float)QROWS;   // q_acc
    out[3] = accum[3] / (float)PROWS;   // p_acc
  }
}

// ---------------- host orchestration ----------------
static inline int cdiv(long a, long b) { return (int)((a + b - 1) / b); }

static void launch_gemm(const _Float16* A, int lda, const _Float16* Bp,
                        const float* bias, float* Cf32, _Float16* Cf16, int ldc,
                        int M, int N, int K, int act, hipStream_t s) {
  long tiles = (long)(M / 32) * (N / 32);
  int blocks = cdiv(tiles * 32, 256);
  k_gemm<<<blocks, 256, 0, s>>>(A, lda, Bp, bias, Cf32, Cf16, ldc, M, N, K, act);
}

static void launch_pack(_Float16* dst, const float* src, int K, int N, hipStream_t s) {
  long total = (long)(K / 32) * (N / 16) * 32;
  k_pack_b<<<cdiv(total, 256), 256, 0, s>>>(dst, src, K, N);
}

extern "C" void kernel_launch(void* const* d_in, const int* in_sizes, int n_in,
                              void* d_out, int out_size, void* d_ws, size_t ws_size,
                              hipStream_t stream) {
  // ---- inputs (setup_inputs order) ----
  const int*  step_eid   = (const int*)d_in[0];
  const int*  step_pred  = (const int*)d_in[1];
  const unsigned char* step_pmask = (const unsigned char*)d_in[2];
  const int*  step_v     = (const int*)d_in[3];
  const int*  step_in0   = (const int*)d_in[4];
  const unsigned char* step_m0 = (const unsigned char*)d_in[5];
  const int*  step_in1   = (const int*)d_in[6];
  const unsigned char* step_m1 = (const unsigned char*)d_in[7];
  const int*  step_label = (const int*)d_in[8];
  const int*  edge_src   = (const int*)d_in[9];
  const int*  edge_dst   = (const int*)d_in[10];
  const int*  root_ids   = (const int*)d_in[11];
  const int*  wid        = (const int*)d_in[12];
  const float* tree_vec  = (const float*)d_in[13];
  const float* embedding = (const float*)d_in[14];
  const float* Wz = (const float*)d_in[15]; const float* bz = (const float*)d_in[16];
  const float* Wr = (const float*)d_in[17]; const float* Ur = (const float*)d_in[18];
  const float* br = (const float*)d_in[19];
  const float* Wh = (const float*)d_in[20]; const float* bh = (const float*)d_in[21];
  const float* W_w = (const float*)d_in[22]; const float* W_b = (const float*)d_in[23];
  const float* U_w = (const float*)d_in[24]; const float* U_b = (const float*)d_in[25];
  const float* Wo_w = (const float*)d_in[26]; const float* Wo_b = (const float*)d_in[27];
  const float* Us_w = (const float*)d_in[28]; const float* Us_b = (const float*)d_in[29];
  (void)in_sizes; (void)n_in; (void)out_size; (void)ws_size;

  // ---- workspace layout ----
  char* base = (char*)d_ws;
  size_t off = 0;
  auto alloc = [&](size_t bytes) -> char* {
    char* p = base + off;
    off = (off + bytes + 255) & ~(size_t)255;
    return p;
  };
  _Float16* xf16   = (_Float16*)alloc((size_t)BB * NN * HH * 2);       // 12.6 MB
  _Float16* WzP    = (_Float16*)alloc((size_t)2 * HH * HH * 2);        // packed B
  _Float16* WhP    = (_Float16*)alloc((size_t)2 * HH * HH * 2);
  _Float16* W3P    = (_Float16*)alloc((size_t)2 * HH * HH * 2);        // [Wr; Ur] packed
  _Float16* WwP    = (_Float16*)alloc((size_t)(HH + LL) * HH * 2);
  _Float16* UwP    = (_Float16*)alloc((size_t)(2 * HH + LL) * HH * 2);
  _Float16* WoP    = (_Float16*)alloc((size_t)HH * VV * 2);
  _Float16* tv16   = (_Float16*)alloc((size_t)BB * LL * 2);
  _Float16* m16    = (_Float16*)alloc((size_t)ETOT * HH * 2);          // 24.6 MB
  _Float16* rm16   = (_Float16*)alloc((size_t)ETOT * HH * 2);          // 24.6 MB
  _Float16* hs     = (_Float16*)alloc((size_t)TT * BB * HH * 2);       // 24.6 MB
  _Float16* A1     = (_Float16*)alloc((size_t)BB * 512 * 2);
  _Float16* A2     = (_Float16*)alloc((size_t)BB * 512 * 2);
  _Float16* A3     = (_Float16*)alloc((size_t)BB * 512 * 2);
  float*    zpre   = (float*)alloc((size_t)BB * HH * 4);
  float*    mtpre  = (float*)alloc((size_t)BB * HH * 4);
  float*    rpre   = (float*)alloc((size_t)BB * HH * 4);
  _Float16* relu1q = (_Float16*)alloc((size_t)QROWS * HH * 2);         // 12.6 MB
  _Float16* relu1p = (_Float16*)alloc((size_t)PROWS * HH * 2);         // 24.9 MB
  int*      pnode  = (int*)alloc((size_t)PROWS * 4);
  int*      ph     = (int*)alloc((size_t)PROWS * 4);
  int*      pb     = (int*)alloc((size_t)PROWS * 4);
  int*      qh     = (int*)alloc((size_t)QROWS * 4);
  float*    accum  = (float*)alloc(8 * 4);

  // ---- init: pack weights into WMMA-native fragments, gather embeddings ----
  k_zero_f32<<<1, 8, 0, stream>>>(accum, 8);
  launch_pack(WzP, Wz, 2 * HH, HH, stream);
  launch_pack(WhP, Wh, 2 * HH, HH, stream);
  launch_pack(W3P, Wr, HH, HH, stream);                       // kt 0..7
  launch_pack(W3P + (size_t)8 * HH * 32, Ur, HH, HH, stream); // kt 8..15
  launch_pack(WwP, W_w, HH + LL, HH, stream);
  launch_pack(UwP, U_w, 2 * HH + LL, HH, stream);
  launch_pack(WoP, Wo_w, HH, VV, stream);
  k_cvt<<<cdiv(BB * LL, 256), 256, 0, stream>>>(tv16, tree_vec, BB * LL);
  k_embed<<<cdiv((long)BB * NN * HH, 256), 256, 0, stream>>>(embedding, wid, xf16, BB * NN * HH);
  k_zero_h<<<cdiv((long)ETOT * HH, 256), 256, 0, stream>>>(m16, ETOT * HH);
  k_zero_h<<<cdiv((long)ETOT * HH, 256), 256, 0, stream>>>(rm16, ETOT * HH);
  k_build_idx<<<cdiv(PROWS, 256), 256, 0, stream>>>(root_ids, step_v, pnode, ph, pb, qh);

  // ---- sequential GRU scan over line-graph steps ----
  const int elemBlocks = cdiv(BB * HH, 256);
  for (int t = 0; t < TT; ++t) {
    const int* eid_t  = step_eid  + t * BB;
    const int* pred_t = step_pred + t * BB;
    const unsigned char* pm_t = step_pmask + t * BB;
    k_step_prepare<<<elemBlocks, 256, 0, stream>>>(eid_t, pred_t, pm_t, edge_src,
                                                   xf16, m16, rm16, A1, A2);
    launch_gemm(A1, 512, WzP, nullptr, zpre,  nullptr, HH, BB, HH, 512, 0, stream);
    launch_gemm(A2, 512, WhP, nullptr, mtpre, nullptr, HH, BB, HH, 512, 0, stream);
    k_step_elem<<<elemBlocks, 256, 0, stream>>>(zpre, mtpre, bz, bh, A1, eid_t, edge_dst,
                                                xf16, m16, A3);
    launch_gemm(A3, 512, W3P, nullptr, rpre, nullptr, HH, BB, HH, 512, 0, stream);
    k_step_finish<<<elemBlocks, 256, 0, stream>>>(rpre, br, eid_t,
                                                  step_in0 + t * BB, step_m0 + t * BB,
                                                  step_in1 + t * BB, step_m1 + t * BB,
                                                  m16, rm16, hs + (size_t)t * BB * HH);
  }

  // ---- q head: relu1q = relu([h | tv] @ W_w + W_b)   (K = 256 + 64) ----
  {
    long tiles = (long)(QROWS / 32) * (HH / 32);
    k_gemm_concat<<<cdiv(tiles * 32, 256), 256, 0, stream>>>(
        hs, qh, HH, HH,
        tv16, pb, LL, LL,
        nullptr, nullptr, 0, 0,
        WwP, W_b, relu1q, HH, QROWS, HH, 1);
  }
  // ---- p head: relu1p = relu([x_v | h | tv] @ U_w + U_b)  (K = 256+256+64) ----
  {
    long tiles = (long)(PROWS / 32) * (HH / 32);
    k_gemm_concat<<<cdiv(tiles * 32, 256), 256, 0, stream>>>(
        xf16, pnode, HH, HH,
        hs, ph, HH, HH,
        tv16, pb, LL, LL,
        UwP, U_b, relu1p, HH, PROWS, HH, 1);
  }

  // ---- q logits + log-softmax loss/acc (WMMA into LDS, per-row reduce) ----
  k_qsoftmax<<<QROWS / 16, 256, 0, stream>>>(relu1q, WoP, Wo_b, wid, root_ids, step_v, accum);
  // ---- p scalar logits + BCE loss/acc ----
  k_ploss<<<cdiv(PROWS, 8), 256, 0, stream>>>(relu1p, Us_w, Us_b, step_label, accum);

  k_final<<<1, 32, 0, stream>>>(accum, (float*)d_out);
}
